// MultiHeadSelfAttention_12859132084421
// MI455X (gfx1250) — matrix operand order repaired and checked by reading, not compiled, not run on hardware
//
#include <hip/hip_runtime.h>
#include <stdint.h>

// ---------------------------------------------------------------------------
// MHA forward for MI455X (gfx1250, wave32).
// bf16 WMMA (v_wmma_f32_16x16x32_bf16) everywhere, fp32 accumulate.
// Attention K/V tiles double-buffered through LDS with async copies
// (GLOBAL_LOAD_ASYNC_TO_LDS_B128 + s_wait_asynccnt) when available.
// ---------------------------------------------------------------------------

typedef __bf16 v16bf __attribute__((ext_vector_type(16)));
typedef float v8f __attribute__((ext_vector_type(8)));
typedef uint32_t u32x4 __attribute__((ext_vector_type(4)));
typedef int v4i_t __attribute__((ext_vector_type(4)));

typedef __attribute__((address_space(1))) v4i_t* gbl_v4i_p;
typedef __attribute__((address_space(3))) v4i_t* lds_v4i_p;

union Frag {
  v16bf v;
  u32x4 q[2];
};

__device__ __forceinline__ uint16_t f2bf(float f) {
  uint32_t u = __builtin_bit_cast(uint32_t, f);
  uint32_t r = u + 0x7FFFu + ((u >> 16) & 1u);  // round-to-nearest-even
  return (uint16_t)(r >> 16);
}

__device__ __forceinline__ v8f wmma_bf16(const Frag& a, const Frag& b, v8f c) {
  return __builtin_amdgcn_wmma_f32_16x16x32_bf16(
      /*neg_a=*/false, a.v, /*neg_b=*/false, b.v,
      /*c_mod=*/(short)0, c, /*reuse_a=*/false, /*reuse_b=*/false);
}

// Reductions across 16-lane halves (xor masks < 16 never cross the half).
__device__ __forceinline__ float rmax16(float v) {
#pragma unroll
  for (int m = 1; m < 16; m <<= 1) v = fmaxf(v, __shfl_xor(v, m, 32));
  return v;
}
__device__ __forceinline__ float rsum16(float v) {
#pragma unroll
  for (int m = 1; m < 16; m <<= 1) v += __shfl_xor(v, m, 32);
  return v;
}

#if __has_builtin(__builtin_amdgcn_global_load_async_to_lds_b128)
#define HAVE_ASYNC_LDS 1
__device__ __forceinline__ void async_cp16(const uint16_t* g, uint16_t* l) {
  __builtin_amdgcn_global_load_async_to_lds_b128(
      (gbl_v4i_p)(g), (lds_v4i_p)(l), /*offset=*/0, /*cpol=*/0);
}
#else
#define HAVE_ASYNC_LDS 0
#endif

// Problem constants
constexpr int S_LEN  = 2048;
constexpr int DMODEL = 1024;
constexpr int NHEAD  = 16;
constexpr int DK     = 64;
constexpr int BATCH  = 2;
constexpr int MROWS  = BATCH * S_LEN;  // 4096

// ---------------------------------------------------------------------------
// fp32 -> bf16 elementwise convert (vec4)
// ---------------------------------------------------------------------------
__global__ void cvt_bf16_kernel(const float* __restrict__ x,
                                uint16_t* __restrict__ y, int n) {
  int i = (blockIdx.x * blockDim.x + threadIdx.x) * 4;
  if (i + 3 < n) {
    float4 f = *(const float4*)(x + i);
    y[i + 0] = f2bf(f.x);
    y[i + 1] = f2bf(f.y);
    y[i + 2] = f2bf(f.z);
    y[i + 3] = f2bf(f.w);
  }
}

// ---------------------------------------------------------------------------
// Transpose-convert: W[in=1024][out=1024] fp32 -> Wt[out][in] bf16.
// ---------------------------------------------------------------------------
__global__ void transpose_bf16_kernel(const float* __restrict__ W,
                                      uint16_t* __restrict__ Wt) {
  __shared__ float tile[32][33];
  const int x = blockIdx.x * 32 + threadIdx.x;
  const int y0 = blockIdx.y * 32 + threadIdx.y;
#pragma unroll
  for (int j = 0; j < 32; j += 8)
    tile[threadIdx.y + j][threadIdx.x] = W[(size_t)(y0 + j) * DMODEL + x];
  __syncthreads();
  const int x2 = blockIdx.y * 32 + threadIdx.x;
  const int y2 = blockIdx.x * 32 + threadIdx.y;
#pragma unroll
  for (int j = 0; j < 32; j += 8)
    Wt[(size_t)(y2 + j) * DMODEL + x2] = f2bf(tile[threadIdx.x][threadIdx.y + j]);
}

// ---------------------------------------------------------------------------
// Wave-level GEMM: C[4096x1024] = A[4096x1024] * Bt^T + bias.
// Each wave computes a 16(M) x 64(N) tile: 4 accumulators, 4 WMMAs / K-step.
// MODE 0: -> Q  [B,H,S,Dk] bf16, scaled by 1/sqrt(Dk)
// MODE 1: -> K  [B,H,S,Dk] bf16
// MODE 2: -> V  [B,H,Dk,S] bf16 (transposed for PV B-fragments)
// MODE 3: -> out fp32 row-major [M][N]
// ---------------------------------------------------------------------------
template <int MODE>
__global__ __launch_bounds__(128) void gemm_kernel(
    const uint16_t* __restrict__ A, const uint16_t* __restrict__ Bt,
    const float* __restrict__ bias, void* __restrict__ outp) {
  const int lane  = threadIdx.x & 31;
  const int wave  = threadIdx.x >> 5;
  const int lmod  = lane & 15;
  const int lhalf = lane >> 4;

  const int m0 = blockIdx.y * 16;
  const int n0 = (blockIdx.x * 4 + wave) * 64;

  v8f acc[4] = {};

  const uint16_t* arow  = A + (size_t)(m0 + lmod) * DMODEL + 16 * lhalf;
  const uint16_t* brow0 = Bt + (size_t)(n0 + lmod) * DMODEL + 16 * lhalf;

  for (int kk = 0; kk < DMODEL; kk += 32) {
    __builtin_prefetch(arow + kk + 256, 0, 1);  // global_prefetch_b8
    Frag a;
    a.q[0] = *(const u32x4*)(arow + kk);
    a.q[1] = *(const u32x4*)(arow + kk + 8);
#pragma unroll
    for (int j = 0; j < 4; ++j) {
      const uint16_t* bp = brow0 + (size_t)j * 16 * DMODEL + kk;
      Frag bb;
      bb.q[0] = *(const u32x4*)(bp);
      bb.q[1] = *(const u32x4*)(bp + 8);
      acc[j] = wmma_bf16(a, bb, acc[j]);
    }
  }

  // Epilogue. C layout: VGPR r -> row m0 + r + 8*lhalf, col n0 + j*16 + lmod.
#pragma unroll
  for (int j = 0; j < 4; ++j) {
    const int n = n0 + j * 16 + lmod;
    const float bn = bias[n];
#pragma unroll
    for (int r = 0; r < 8; ++r) {
      const int m = m0 + r + 8 * lhalf;
      float val = acc[j][r] + bn;
      if (MODE == 0 || MODE == 1) {
        if (MODE == 0) val *= 0.125f;  // 1/sqrt(64) folded into Q
        const int b = m >> 11, s = m & 2047;
        const int h = n >> 6, d = n & 63;
        ((uint16_t*)outp)[((size_t)(b * NHEAD + h) * S_LEN + s) * DK + d] =
            f2bf(val);
      } else if (MODE == 2) {
        const int b = m >> 11, s = m & 2047;
        const int h = n >> 6, d = n & 63;
        ((uint16_t*)outp)[((size_t)(b * NHEAD + h) * DK + d) * S_LEN + s] =
            f2bf(val);
      } else {
        ((float*)outp)[(size_t)m * DMODEL + n] = val;
      }
    }
  }
}

// ---------------------------------------------------------------------------
// Flash attention: one wave per (b, h, 16-query tile).
// K/V tiles (32 keys) staged in LDS via async copies, double buffered:
//   16 GLOBAL_LOAD_ASYNC_TO_LDS_B128 per tile per wave (8 K-chunks, 8 V-chunks
//   of 16 B per lane), synchronized with s_wait_asynccnt only.
// Scores = 4 WMMAs; online softmax via 16-lane shuffle reductions; P re-laid
// C->A through a 1 KB LDS tile; PV = 4 WMMAs.
// ---------------------------------------------------------------------------
__global__ __launch_bounds__(128) void attn_kernel(
    const uint16_t* __restrict__ Q, const uint16_t* __restrict__ Kb,
    const uint16_t* __restrict__ Vt, uint16_t* __restrict__ Oa) {
  // stage[wave][slot][chunk 0..15][lane][8 x bf16 = 16B]  -> 64 KB total
  __shared__ __align__(16) uint16_t stage[4][2][16][32][8];
  __shared__ __align__(16) uint16_t pbuf[4][16][32];

  const int lane  = threadIdx.x & 31;
  const int wave  = threadIdx.x >> 5;
  const int lmod  = lane & 15;
  const int lhalf = lane >> 4;

  const int b  = blockIdx.z;
  const int h  = blockIdx.y;
  const int qt = blockIdx.x * 4 + wave;

  const size_t bh = (size_t)(b * NHEAD + h) * S_LEN * DK;
  const uint16_t* Qh = Q + bh;
  const uint16_t* Kh = Kb + bh;
  const uint16_t* Vh = Vt + bh;  // [Dk][S] per (b,h)

  // Q A-fragments for d 0..31 (aq0) and 32..63 (aq1); loaded once.
  const uint16_t* qrow = Qh + (size_t)(qt * 16 + lmod) * DK + 16 * lhalf;
  Frag aq0, aq1;
  aq0.q[0] = *(const u32x4*)(qrow);
  aq0.q[1] = *(const u32x4*)(qrow + 8);
  aq1.q[0] = *(const u32x4*)(qrow + 32);
  aq1.q[1] = *(const u32x4*)(qrow + 40);

#if HAVE_ASYNC_LDS
  // Issue the 16 async 16B copies for key-tile kt into LDS slot `slot`.
  auto issue_tile = [&](int kt, int slot) {
    const int key0 = kt * 32;
#pragma unroll
    for (int kh = 0; kh < 2; ++kh) {
      const uint16_t* krow =
          Kh + (size_t)(key0 + kh * 16 + lmod) * DK + 16 * lhalf;
#pragma unroll
      for (int d0 = 0; d0 < 2; ++d0)
#pragma unroll
        for (int j = 0; j < 2; ++j)
          async_cp16(krow + d0 * 32 + j * 8,
                     &stage[wave][slot][(d0 * 2 + kh) * 2 + j][lane][0]);
    }
#pragma unroll
    for (int t = 0; t < 4; ++t) {
      const uint16_t* vrow =
          Vh + (size_t)(t * 16 + lmod) * S_LEN + key0 + 16 * lhalf;
#pragma unroll
      for (int j = 0; j < 2; ++j)
        async_cp16(vrow + j * 8, &stage[wave][slot][8 + t * 2 + j][lane][0]);
    }
  };
#endif

  float mrow[8], lrow[8];
#pragma unroll
  for (int r = 0; r < 8; ++r) { mrow[r] = -1e30f; lrow[r] = 0.0f; }
  v8f acc[4] = {};

#if HAVE_ASYNC_LDS
  issue_tile(0, 0);
#endif

  for (int kt = 0; kt < S_LEN / 32; ++kt) {
    const int key0 = kt * 32;
    Frag bk[2][2], bv[4];

#if HAVE_ASYNC_LDS
    const int slot = kt & 1;
    if (kt + 1 < S_LEN / 32) {
      issue_tile(kt + 1, slot ^ 1);
      asm volatile("s_wait_asynccnt 0x10" ::: "memory");  // tile kt landed
    } else {
      asm volatile("s_wait_asynccnt 0x0" ::: "memory");
    }
    // Fragments from the LDS stage (ds_load_b128).
#pragma unroll
    for (int d0 = 0; d0 < 2; ++d0)
#pragma unroll
      for (int kh = 0; kh < 2; ++kh)
#pragma unroll
        for (int j = 0; j < 2; ++j)
          bk[d0][kh].q[j] =
              *(const u32x4*)&stage[wave][slot][(d0 * 2 + kh) * 2 + j][lane][0];
#pragma unroll
    for (int t = 0; t < 4; ++t)
#pragma unroll
      for (int j = 0; j < 2; ++j)
        bv[t].q[j] = *(const u32x4*)&stage[wave][slot][8 + t * 2 + j][lane][0];
#else
    // Direct global fragment loads (fallback).
#pragma unroll
    for (int kh = 0; kh < 2; ++kh) {
      const uint16_t* krow =
          Kh + (size_t)(key0 + kh * 16 + lmod) * DK + 16 * lhalf;
#pragma unroll
      for (int d0 = 0; d0 < 2; ++d0) {
        bk[d0][kh].q[0] = *(const u32x4*)(krow + d0 * 32);
        bk[d0][kh].q[1] = *(const u32x4*)(krow + d0 * 32 + 8);
      }
    }
#pragma unroll
    for (int t = 0; t < 4; ++t) {
      const uint16_t* vrow =
          Vh + (size_t)(t * 16 + lmod) * S_LEN + key0 + 16 * lhalf;
      bv[t].q[0] = *(const u32x4*)(vrow);
      bv[t].q[1] = *(const u32x4*)(vrow + 8);
    }
#endif

    v8f s0 = {}, s1 = {};
    s0 = wmma_bf16(aq0, bk[0][0], s0);
    s0 = wmma_bf16(aq1, bk[1][0], s0);
    s1 = wmma_bf16(aq0, bk[0][1], s1);
    s1 = wmma_bf16(aq1, bk[1][1], s1);

    // Online softmax. Row M = r + 8*lhalf lives across the 16-lane half.
#pragma unroll
    for (int r = 0; r < 8; ++r) {
      const float v0 = s0[r], v1 = s1[r];
      float tmax = rmax16(fmaxf(v0, v1));
      const float mnew  = fmaxf(mrow[r], tmax);
      const float scale = __expf(mrow[r] - mnew);
      const float p0 = __expf(v0 - mnew);
      const float p1 = __expf(v1 - mnew);
      lrow[r] = lrow[r] * scale + rsum16(p0 + p1);
      mrow[r] = mnew;
#pragma unroll
      for (int t = 0; t < 4; ++t) acc[t][r] *= scale;
      const int prow = r + 8 * lhalf;
      pbuf[wave][prow][lmod]      = f2bf(p0);
      pbuf[wave][prow][lmod + 16] = f2bf(p1);
    }
    asm volatile("" ::: "memory");  // keep ds stores before ds loads

    // P as A-fragment (16 queries x 32 keys) from LDS.
    Frag pa;
    const uint16_t* pp = &pbuf[wave][lmod][16 * lhalf];
    pa.q[0] = *(const u32x4*)(pp);
    pa.q[1] = *(const u32x4*)(pp + 8);

#pragma unroll
    for (int t = 0; t < 4; ++t) acc[t] = wmma_bf16(pa, bv[t], acc[t]);
    asm volatile("" ::: "memory");  // pbuf reads precede next iter's writes
  }

  // Normalize and store bf16 into [B,S,DMODEL] layout.
  const int qbase = qt * 16;
#pragma unroll
  for (int t = 0; t < 4; ++t) {
    const int d = t * 16 + lmod;
#pragma unroll
    for (int r = 0; r < 8; ++r) {
      const int q = qbase + r + 8 * lhalf;
      const float val = acc[t][r] / lrow[r];
      Oa[((size_t)(b * S_LEN + q)) * DMODEL + h * DK + d] = f2bf(val);
    }
  }
}

// ---------------------------------------------------------------------------
// Host launcher
// ---------------------------------------------------------------------------
extern "C" void kernel_launch(void* const* d_in, const int* in_sizes, int n_in,
                              void* d_out, int out_size, void* d_ws,
                              size_t ws_size, hipStream_t stream) {
  const float* x  = (const float*)d_in[0];
  const float* Wq = (const float*)d_in[1];
  const float* bq = (const float*)d_in[2];
  const float* Wk = (const float*)d_in[3];
  const float* bk = (const float*)d_in[4];
  const float* Wv = (const float*)d_in[5];
  const float* bv = (const float*)d_in[6];
  const float* Wo = (const float*)d_in[7];
  const float* bo = (const float*)d_in[8];
  float* out = (float*)d_out;

  char* ws = (char*)d_ws;
  const size_t MB = 1u << 20;
  uint16_t* xb  = (uint16_t*)(ws);             //  8 MB: x bf16 [4096][1024]
  uint16_t* Wqt = (uint16_t*)(ws + 8 * MB);    //  2 MB each, [out][in] bf16
  uint16_t* Wkt = Wqt + DMODEL * DMODEL;
  uint16_t* Wvt = Wkt + DMODEL * DMODEL;
  uint16_t* Wot = Wvt + DMODEL * DMODEL;
  uint16_t* Qb  = (uint16_t*)(ws + 16 * MB);   //  8 MB [B,H,S,Dk]
  uint16_t* Kbf = (uint16_t*)(ws + 24 * MB);   //  8 MB [B,H,S,Dk]
  uint16_t* Vtb = (uint16_t*)(ws + 32 * MB);   //  8 MB [B,H,Dk,S]
  uint16_t* Ob  = (uint16_t*)(ws + 40 * MB);   //  8 MB [B,S,DMODEL]

  {
    const int n = MROWS * DMODEL;
    cvt_bf16_kernel<<<n / (256 * 4), 256, 0, stream>>>(x, xb, n);
  }
  {
    dim3 g(DMODEL / 32, DMODEL / 32), blk(32, 8);
    transpose_bf16_kernel<<<g, blk, 0, stream>>>(Wq, Wqt);
    transpose_bf16_kernel<<<g, blk, 0, stream>>>(Wk, Wkt);
    transpose_bf16_kernel<<<g, blk, 0, stream>>>(Wv, Wvt);
    transpose_bf16_kernel<<<g, blk, 0, stream>>>(Wo, Wot);
  }
  {
    dim3 g(DMODEL / 256, MROWS / 16);  // (4, 256); 4 waves x 64 cols per block
    gemm_kernel<0><<<g, 128, 0, stream>>>(xb, Wqt, bq, Qb);
    gemm_kernel<1><<<g, 128, 0, stream>>>(xb, Wkt, bk, Kbf);
    gemm_kernel<2><<<g, 128, 0, stream>>>(xb, Wvt, bv, Vtb);
  }
  {
    dim3 g(S_LEN / 16 / 4, NHEAD, BATCH);  // (32, 16, 2)
    attn_kernel<<<g, 128, 0, stream>>>(Qb, Kbf, Vtb, Ob);
  }
  {
    dim3 g(DMODEL / 256, MROWS / 16);
    gemm_kernel<3><<<g, 128, 0, stream>>>(Ob, Wot, bo, out);
  }
}
